// AttentionHead_35639638622747
// MI455X (gfx1250) — matrix operand order repaired and checked by reading, not compiled, not run on hardware
//
#include <hip/hip_runtime.h>
#include <hip/hip_bf16.h>

// ---- problem constants (from reference) ----
constexpr int NB    = 2;      // batch
constexpr int SEQ   = 4096;   // sequence length
constexpr int DIM   = 512;    // model dim
constexpr int HDIM  = 64;     // head dim
constexpr int WSC   = 128;    // SCALE; window = [-128, 127]
constexpr int WCOLS = 288;    // 272 union-window cols padded to 9*32

typedef __attribute__((ext_vector_type(16))) __bf16 v16bf;
typedef __attribute__((ext_vector_type(8)))  __bf16 bf16x8;
typedef __attribute__((ext_vector_type(8)))  float  v8f;

static __device__ __forceinline__ v16bf cat16(bf16x8 lo, bf16x8 hi) {
    return __builtin_shufflevector(lo, hi, 0,1,2,3,4,5,6,7,8,9,10,11,12,13,14,15);
}

// ---------------------------------------------------------------------------
// Kernel 0: transpose + convert weights to bf16: wt[(mat*64 + n)*512 + k]
// ---------------------------------------------------------------------------
__global__ __launch_bounds__(256) void prep_w_kernel(
    const float* __restrict__ wq, const float* __restrict__ wk,
    const float* __restrict__ wv, __bf16* __restrict__ wt)
{
    int i = blockIdx.x * 256 + threadIdx.x;          // over 3*64*512
    if (i >= 3 * HDIM * DIM) return;
    int mat = i / (HDIM * DIM);
    int r   = i % (HDIM * DIM);
    int n   = r / DIM;
    int k   = r % DIM;
    const float* w = (mat == 0) ? wq : ((mat == 1) ? wk : wv);
    wt[i] = (__bf16)w[k * HDIM + n];
}

// ---------------------------------------------------------------------------
// Kernel 1: q,k,v = x @ W; fp32 in (b128 loads), bf16 out.
// B tiles preloaded in groups of 6 so each load batch is covered by 6 WMMAs.
// q,k row-major; v stored transposed vt[b][n][s].
// ---------------------------------------------------------------------------
__global__ __launch_bounds__(64) void proj_qkv_kernel(
    const float* __restrict__ x, const __bf16* __restrict__ wt,
    __bf16* __restrict__ qb, __bf16* __restrict__ kb, __bf16* __restrict__ vt)
{
    const int lane  = threadIdx.x & 31;
    const int wave  = threadIdx.x >> 5;
    const int mt    = blockIdx.x * 2 + wave;   // 512 M-tiles over 8192 rows
    const int hlf   = lane >> 4;
    const int mlane = lane & 15;

    const long rowA = (long)mt * 16 + mlane;

    v8f acc[12];
    v8f zero = {0.f,0.f,0.f,0.f,0.f,0.f,0.f,0.f};
#pragma unroll
    for (int t = 0; t < 12; ++t) acc[t] = zero;

    for (int ks = 0; ks < DIM / 32; ++ks) {
        // A tile 16x32: two groups of 8 contiguous f32 per lane
        const float4* ap =
            reinterpret_cast<const float4*>(x + rowA * DIM + ks * 32);
        float4 g0a = ap[hlf * 2];
        float4 g0b = ap[hlf * 2 + 1];
        float4 g1a = ap[4 + hlf * 2];
        float4 g1b = ap[5 + hlf * 2];
        v16bf a;
        a[0]=(__bf16)g0a.x; a[1]=(__bf16)g0a.y; a[2]=(__bf16)g0a.z; a[3]=(__bf16)g0a.w;
        a[4]=(__bf16)g0b.x; a[5]=(__bf16)g0b.y; a[6]=(__bf16)g0b.z; a[7]=(__bf16)g0b.w;
        a[8]=(__bf16)g1a.x; a[9]=(__bf16)g1a.y; a[10]=(__bf16)g1a.z; a[11]=(__bf16)g1a.w;
        a[12]=(__bf16)g1b.x; a[13]=(__bf16)g1b.y; a[14]=(__bf16)g1b.z; a[15]=(__bf16)g1b.w;

        // Two pipelined half-batches of 6 B tiles each
#pragma unroll
        for (int g = 0; g < 2; ++g) {
            v16bf bm[6];
#pragma unroll
            for (int u = 0; u < 6; ++u) {
                const int t = g * 6 + u;
                const __bf16* wrow =
                    wt + ((long)(t >> 2) * HDIM + (t & 3) * 16 + mlane) * DIM
                       + ks * 32 + hlf * 8;
                bm[u] = cat16(*reinterpret_cast<const bf16x8*>(wrow),
                              *reinterpret_cast<const bf16x8*>(wrow + 16));
            }
#pragma unroll
            for (int u = 0; u < 6; ++u) {
                const int t = g * 6 + u;
                acc[t] = __builtin_amdgcn_wmma_f32_16x16x32_bf16(
                    false, a, false, bm[u], (short)0, acc[t], false, false);
            }
        }
    }

    // C layout: lane element r -> (m = r + 8*hlf, n = mlane)
#pragma unroll
    for (int t = 0; t < 8; ++t) {              // q, k: row-major bf16
        __bf16* o = (t < 4) ? qb : kb;
        const int nt = t & 3;
#pragma unroll
        for (int r = 0; r < 8; ++r) {
            int m = r + 8 * hlf;
            o[((long)mt * 16 + m) * HDIM + nt * 16 + mlane] = (__bf16)acc[t][r];
        }
    }
#pragma unroll
    for (int t = 8; t < 12; ++t) {             // v: transposed vt[b][n][s]
        const int nt = t & 3;
#pragma unroll
        for (int r = 0; r < 8; ++r) {
            int  m = r + 8 * hlf;
            long g = (long)mt * 16 + m;        // global row in [0, 8192)
            int  bidx = (int)(g >> 12);        // / SEQ
            int  s    = (int)(g & (SEQ - 1));
            vt[((long)bidx * HDIM + nt * 16 + mlane) * SEQ + s] = (__bf16)acc[t][r];
        }
    }
}

// ---------------------------------------------------------------------------
// Kernel 2: sliding-window attention, one 16-query tile per wave.
// K tiles double-buffered across the kt loop; V tiles preloaded per k-step.
// ---------------------------------------------------------------------------
__global__ __launch_bounds__(64) void swin_attn_kernel(
    const __bf16* __restrict__ qb, const __bf16* __restrict__ kb,
    const __bf16* __restrict__ vt, float* __restrict__ out)
{
    __shared__ float sc[2][16 * WCOLS];
    __shared__ float rowsum[2][16];

    const int lane  = threadIdx.x & 31;
    const int wave  = threadIdx.x >> 5;
    const int tid   = blockIdx.x * 2 + wave;   // 512 tiles = NB * SEQ/16
    const int b     = tid >> 8;
    const int qt    = tid & 255;
    const int s0    = qt * 16;
    const int hlf   = lane >> 4;
    const int mlane = lane & 15;

    float* lsc = sc[wave];

    v16bf zv;
#pragma unroll
    for (int e = 0; e < 16; ++e) zv[e] = (__bf16)0.f;

    // ---- Q as two A tiles; contiguous b128 loads per lane ----
    const __bf16* qrowp = qb + (long)(b * SEQ + s0 + mlane) * HDIM;
    v16bf a0 = cat16(*reinterpret_cast<const bf16x8*>(qrowp + hlf * 8),
                     *reinterpret_cast<const bf16x8*>(qrowp + 16 + hlf * 8));
    v16bf a1 = cat16(*reinterpret_cast<const bf16x8*>(qrowp + 32 + hlf * 8),
                     *reinterpret_cast<const bf16x8*>(qrowp + 48 + hlf * 8));

    // zero pad columns 272..287 (probabilities must be 0 there)
    for (int i = lane; i < 16 * 16; i += 32)
        lsc[(i >> 4) * WCOLS + 272 + (i & 15)] = 0.f;

    // ---- phase 1: scores = Q K^T / 8, masked, into LDS (double-buffered) ---
    auto loadK = [&](int kt, v16bf& b0, v16bf& b1) {
        const int j = s0 - WSC + kt * 16 + mlane;
        const bool inseq = (j >= 0) && (j < SEQ);
        const __bf16* kr = kb + (long)(b * SEQ + (inseq ? j : 0)) * HDIM;
        b0 = cat16(*reinterpret_cast<const bf16x8*>(kr + hlf * 8),
                   *reinterpret_cast<const bf16x8*>(kr + 16 + hlf * 8));
        b1 = cat16(*reinterpret_cast<const bf16x8*>(kr + 32 + hlf * 8),
                   *reinterpret_cast<const bf16x8*>(kr + 48 + hlf * 8));
        if (!inseq) { b0 = zv; b1 = zv; }
    };

    v16bf b0c, b1c, b0n, b1n;
    loadK(0, b0c, b1c);
    for (int kt = 0; kt < 17; ++kt) {
        if (kt < 16) loadK(kt + 1, b0n, b1n);          // prefetch next tile
        const int j = s0 - WSC + kt * 16 + mlane;
        if (kt + 2 < 17 && (unsigned)(j + 32) < (unsigned)SEQ)
            __builtin_prefetch(kb + (long)(b * SEQ + j + 32) * HDIM, 0, 1);

        v8f csc = {0.f,0.f,0.f,0.f,0.f,0.f,0.f,0.f};
        csc = __builtin_amdgcn_wmma_f32_16x16x32_bf16(
            false, a0, false, b0c, (short)0, csc, false, false);
        csc = __builtin_amdgcn_wmma_f32_16x16x32_bf16(
            false, a1, false, b1c, (short)0, csc, false, false);
#pragma unroll
        for (int r = 0; r < 8; ++r) {
            int m   = r + 8 * hlf;
            int rel = j - (s0 + m);
            float v = csc[r] * 0.125f;         // 1/sqrt(64)
            if (rel < -WSC || rel >= WSC) v = -1e30f;   // outside window
            // in-window out-of-seq keys keep score exactly 0 (ref semantics)
            lsc[m * WCOLS + kt * 16 + mlane] = v;
        }
        b0c = b0n; b1c = b1n;
    }
    __syncthreads();

    // ---- phase 2: softmax per row; lanes L and L+16 split 272 cols ----
    {
        const int row = mlane;
        const int c0  = hlf * 136;
        float mx = -1e30f;
        for (int c = 0; c < 136; ++c)
            mx = fmaxf(mx, lsc[row * WCOLS + c0 + c]);
        mx = fmaxf(mx, __shfl_xor(mx, 16, 32));
        float sum = 0.f;
        for (int c = 0; c < 136; ++c) {
            float p = __expf(lsc[row * WCOLS + c0 + c] - mx);
            lsc[row * WCOLS + c0 + c] = p;
            sum += p;
        }
        sum += __shfl_xor(sum, 16, 32);
        if (hlf == 0) rowsum[wave][row] = sum;
    }
    __syncthreads();

    // ---- phase 3: O = P V. P from LDS (b128), V tiles preloaded ----
    v8f acc[4];
    v8f zero = {0.f,0.f,0.f,0.f,0.f,0.f,0.f,0.f};
#pragma unroll
    for (int nt = 0; nt < 4; ++nt) acc[nt] = zero;

    for (int ks = 0; ks < 9; ++ks) {           // 288 union cols / 32
        const int jb = s0 - WSC + ks * 32 + hlf * 16;   // 16 consecutive keys
        // preload 4 V B-tiles first (b128 pairs from transposed V)
        v16bf bm[4];
#pragma unroll
        for (int nt = 0; nt < 4; ++nt) {
            const __bf16* vrow = vt + ((long)b * HDIM + nt * 16 + mlane) * SEQ;
            if (jb >= 0 && jb + 15 < SEQ) {             // fast contiguous path
                bm[nt] = cat16(*reinterpret_cast<const bf16x8*>(vrow + jb),
                               *reinterpret_cast<const bf16x8*>(vrow + jb + 8));
            } else {                                    // sequence boundary
                bm[nt] = zv;
#pragma unroll
                for (int e = 0; e < 16; ++e) {
                    int j = jb + e;
                    if (j >= 0 && j < SEQ) bm[nt][e] = vrow[j];
                }
            }
        }
        // A operand from LDS probabilities (float4 -> ds_load_b128)
        const float4* lp =
            reinterpret_cast<const float4*>(&lsc[mlane * WCOLS + ks * 32]);
        float4 g0a = lp[hlf * 4];
        float4 g0b = lp[hlf * 4 + 1];
        float4 g1a = lp[hlf * 4 + 2];
        float4 g1b = lp[hlf * 4 + 3];
        v16bf a;
        a[0]=(__bf16)g0a.x; a[1]=(__bf16)g0a.y; a[2]=(__bf16)g0a.z; a[3]=(__bf16)g0a.w;
        a[4]=(__bf16)g0b.x; a[5]=(__bf16)g0b.y; a[6]=(__bf16)g0b.z; a[7]=(__bf16)g0b.w;
        a[8]=(__bf16)g1a.x; a[9]=(__bf16)g1a.y; a[10]=(__bf16)g1a.z; a[11]=(__bf16)g1a.w;
        a[12]=(__bf16)g1b.x; a[13]=(__bf16)g1b.y; a[14]=(__bf16)g1b.z; a[15]=(__bf16)g1b.w;

#pragma unroll
        for (int nt = 0; nt < 4; ++nt) {
            acc[nt] = __builtin_amdgcn_wmma_f32_16x16x32_bf16(
                false, a, false, bm[nt], (short)0, acc[nt], false, false);
        }
    }

#pragma unroll
    for (int nt = 0; nt < 4; ++nt) {
#pragma unroll
        for (int r = 0; r < 8; ++r) {
            int m = r + 8 * hlf;
            float inv = 1.f / rowsum[wave][m];
            out[((long)(b * SEQ + s0 + m)) * HDIM + nt * 16 + mlane] =
                acc[nt][r] * inv;
        }
    }
}

// ---------------------------------------------------------------------------
extern "C" void kernel_launch(void* const* d_in, const int* in_sizes, int n_in,
                              void* d_out, int out_size, void* d_ws, size_t ws_size,
                              hipStream_t stream)
{
    const float* x  = (const float*)d_in[0];   // [2,4096,512]
    const float* wq = (const float*)d_in[1];   // [512,64]
    const float* wk = (const float*)d_in[2];
    const float* wv = (const float*)d_in[3];

    const long nrows = (long)NB * SEQ;         // 8192
    __bf16* qb = (__bf16*)d_ws;                // 1 MB
    __bf16* kb = qb + nrows * HDIM;            // 1 MB
    __bf16* vt = kb + nrows * HDIM;            // 1 MB (transposed V)
    __bf16* wt = vt + nrows * HDIM;            // 192 KB (transposed weights)

    prep_w_kernel<<<(3 * HDIM * DIM + 255) / 256, 256, 0, stream>>>(wq, wk, wv, wt);
    proj_qkv_kernel<<<256, 64, 0, stream>>>(x, wt, qb, kb, vt);
    swin_attn_kernel<<<256, 64, 0, stream>>>(qb, kb, vt, (float*)d_out);
}
